// LSTMGenerativeModel_25314537243122
// MI455X (gfx1250) — matrix operand order repaired and checked by reading, not compiled, not run on hardware
//
#include <hip/hip_runtime.h>

typedef _Float16 v16h __attribute__((ext_vector_type(16)));
typedef _Float16 v8h  __attribute__((ext_vector_type(8)));
typedef float    v8f  __attribute__((ext_vector_type(8)));

#define BATCH   8192
#define INPUT   64
#define HIDDEN  512
#define GATES   (4 * HIDDEN)
#define OUTD    64

#define M_TILE  32                    // batch rows per workgroup
#define NWAVES  16
#define THREADS (NWAVES * 32)         // 512 -> 4 waves/SIMD, full VGPR budget
#define NBLOCKS (BATCH / M_TILE)      // 256 workgroups

#define HSTR (HIDDEN + 8)             // 520 halves -> 1040B row stride (16B aligned)
#define ISTR (INPUT + 8)              // 72 halves  -> 144B row stride (16B aligned)

// f16 weight layout inside d_ws (element offsets)
#define WIH_OFF  0
#define WHH_OFF  (GATES * INPUT)                 // 131072
#define WOUT_OFF (WHH_OFF + GATES * HIDDEN)      // 1179648
#define WTOTAL   (WOUT_OFF + OUTD * HIDDEN)      // 1212416 halves

// ---------------------------------------------------------------------------
// fp32 -> f16 weight conversion into workspace (weights are L2-resident after)
// ---------------------------------------------------------------------------
__global__ void lstm_convert_weights(const float* __restrict__ wih,
                                     const float* __restrict__ whh,
                                     const float* __restrict__ wout,
                                     _Float16* __restrict__ ws) {
    int stride = gridDim.x * blockDim.x;
    for (int idx = blockIdx.x * blockDim.x + threadIdx.x; idx < WTOTAL; idx += stride) {
        float v;
        if (idx < WHH_OFF)       v = wih[idx];
        else if (idx < WOUT_OFF) v = whh[idx - WHH_OFF];
        else                     v = wout[idx - WOUT_OFF];
        ws[idx] = (_Float16)v;
    }
}

// ---------------------------------------------------------------------------
// fragment helpers (CDNA5 WMMA 16x16x32 f16, wave32)
// ---------------------------------------------------------------------------
static __device__ __forceinline__ v16h frag_combine(v8h lo, v8h hi) {
    v16h r;
#pragma unroll
    for (int i = 0; i < 8; ++i) { r[i] = lo[i]; r[i + 8] = hi[i]; }
    return r;
}

// A-matrix (16x32 f16): lane m = lane&15; element e -> K = (e&7) + (e>>3)*16 + (lane>>4)*8
static __device__ __forceinline__ v16h load_a_lds(const _Float16* tile, int lane,
                                                  int rstride, int kbase) {
    const _Float16* p = tile + (lane & 15) * rstride + kbase + ((lane >> 4) << 3);
    v8h lo = *(const v8h*)(p);        // K offs +0..7   (ds_load_b128)
    v8h hi = *(const v8h*)(p + 16);   // K offs +16..23 (ds_load_b128)
    return frag_combine(lo, hi);
}

// B-matrix (32x16 f16): lane n = lane&15; element e -> K = (e&7) + (e>>3)*16 + (lane>>4)*8
// B = W^T tile, so lane n reads row (nbase + n) of W, contiguous in K.
// 32-bit element offset -> global_load saddr + voffset form.
static __device__ __forceinline__ v16h load_b_glb(const _Float16* __restrict__ w, int lane,
                                                  int krow, int nbase, int kbase) {
    const int eoff = (nbase + (lane & 15)) * krow + kbase + ((lane >> 4) << 3);
    const _Float16* p = w + eoff;
    v8h lo = *(const v8h*)(p);        // global_load_b128
    v8h hi = *(const v8h*)(p + 16);   // global_load_b128
    return frag_combine(lo, hi);
}

static __device__ __forceinline__ v8f wmma_f16(v16h a, v16h b, v8f c) {
    return __builtin_amdgcn_wmma_f32_16x16x32_f16(false, a, false, b, (short)0, c, false, false);
}

static __device__ __forceinline__ float sig_fast(float x) {
    return 1.0f / (1.0f + __expf(-x));
}
static __device__ __forceinline__ float tanh_fast(float x) {
    return 2.0f * sig_fast(2.0f * x) - 1.0f;
}

// ---------------------------------------------------------------------------
// Persistent LSTM sequence kernel: one WG owns 32 batch rows for all steps.
// 16 waves; wave w owns hidden tiles {2w, 2w+1}; within a tile pass every
// W_hh/W_ih B-fragment is loaded once and used by both 16-row M-tiles.
// h is double-buffered in LDS so the cell update streams straight to LDS.
// ---------------------------------------------------------------------------
__launch_bounds__(THREADS, 1)
__global__ void lstm_gen_kernel(const float* __restrict__ b_lstm,
                                const float* __restrict__ h0,
                                const float* __restrict__ c0,
                                const float* __restrict__ b_out,
                                const _Float16* __restrict__ wbuf,
                                const int* __restrict__ steps_ptr,
                                float* __restrict__ out) {
    __shared__ __align__(16) _Float16 h_lds[2][M_TILE * HSTR]; // double-buffered h (f16)
    __shared__ __align__(16) _Float16 x_lds[M_TILE * ISTR];    // fed-back input (f16)

    const int tid     = threadIdx.x;
    const int lane    = tid & 31;
    const int wave    = tid >> 5;
    const int ncol    = lane & 15;
    const int mhalf   = (lane >> 4) << 3;       // +0 rows for lanes 0-15, +8 for 16-31
    const int rowbase = blockIdx.x * M_TILE;
    const int steps   = steps_ptr[0];

    // ---- initial state: h = h0 broadcast (buffer 0), inp = 0 ----
    for (int idx = tid; idx < M_TILE * HIDDEN; idx += THREADS) {
        int m = idx >> 9, j = idx & (HIDDEN - 1);
        h_lds[0][m * HSTR + j] = (_Float16)h0[j];
    }
    for (int idx = tid; idx < M_TILE * INPUT; idx += THREADS) {
        int m = idx >> 6, j = idx & (INPUT - 1);
        x_lds[m * ISTR + j] = (_Float16)0.0f;
    }

    // c lives in registers in exact D-fragment layout: this lane's column is
    // ncol, rows are r + mhalf within each 16-row M-tile.
    float c_reg[2][2][8];                      // [jj][mt][r]
    float bia[2], bif[2], big[2], bio[2];
#pragma unroll
    for (int jj = 0; jj < 2; ++jj) {
        int jg = (wave * 2 + jj) * 16 + ncol;
        float cv = c0[jg];
#pragma unroll
        for (int mt = 0; mt < 2; ++mt)
#pragma unroll
            for (int r = 0; r < 8; ++r) c_reg[jj][mt][r] = cv;
        bia[jj] = b_lstm[0 * HIDDEN + jg];
        bif[jj] = b_lstm[1 * HIDDEN + jg];
        big[jj] = b_lstm[2 * HIDDEN + jg];
        bio[jj] = b_lstm[3 * HIDDEN + jg];
    }
    const float ob = b_out[(wave & 3) * 16 + ncol];   // phase-2 bias (waves 0..7)

    __syncthreads();

    // Opaque zero offset, redefined every step: keeps the weight pointers'
    // global provenance (-> global_load) but blocks LICM from hoisting the
    // t-invariant W_ih/W_out fragments into (spilled) registers.
    size_t woff = 0;
    int cur = 0;
    for (int t = 0; t < steps; ++t) {
        asm volatile("" : "+s"(woff));
        const _Float16* __restrict__ Wih  = wbuf + (WIH_OFF + woff);
        const _Float16* __restrict__ Whh  = wbuf + (WHH_OFF + woff);
        const _Float16* __restrict__ Wout = wbuf + (WOUT_OFF + woff);

        const _Float16* hcur = h_lds[cur];
        _Float16*       hnxt = h_lds[cur ^ 1];

        // ---------------- phase 1: gates + LSTM cell update ----------------
#pragma unroll
        for (int jj = 0; jj < 2; ++jj) {
            const int jbase = (wave * 2 + jj) * 16;
            v8f acc[2][4];
#pragma unroll
            for (int mt = 0; mt < 2; ++mt)
#pragma unroll
                for (int g = 0; g < 4; ++g) acc[mt][g] = (v8f){};

            // K over inp (64): B-fragments loaded once, used by both M-tiles
            for (int kt = 0; kt < 2; ++kt) {
                const int kb = kt * 32;
                v16h b0 = load_b_glb(Wih, lane, INPUT, 0 * HIDDEN + jbase, kb);
                v16h b1 = load_b_glb(Wih, lane, INPUT, 1 * HIDDEN + jbase, kb);
                v16h b2 = load_b_glb(Wih, lane, INPUT, 2 * HIDDEN + jbase, kb);
                v16h b3 = load_b_glb(Wih, lane, INPUT, 3 * HIDDEN + jbase, kb);
                v16h a0 = load_a_lds(x_lds, lane, ISTR, kb);
                v16h a1 = load_a_lds(x_lds + 16 * ISTR, lane, ISTR, kb);
                acc[0][0] = wmma_f16(a0, b0, acc[0][0]);
                acc[0][1] = wmma_f16(a0, b1, acc[0][1]);
                acc[0][2] = wmma_f16(a0, b2, acc[0][2]);
                acc[0][3] = wmma_f16(a0, b3, acc[0][3]);
                acc[1][0] = wmma_f16(a1, b0, acc[1][0]);
                acc[1][1] = wmma_f16(a1, b1, acc[1][1]);
                acc[1][2] = wmma_f16(a1, b2, acc[1][2]);
                acc[1][3] = wmma_f16(a1, b3, acc[1][3]);
            }
            // K over h (512)
            for (int kt = 0; kt < 16; ++kt) {
                const int kb = kt * 32;
                v16h b0 = load_b_glb(Whh, lane, HIDDEN, 0 * HIDDEN + jbase, kb);
                v16h b1 = load_b_glb(Whh, lane, HIDDEN, 1 * HIDDEN + jbase, kb);
                v16h b2 = load_b_glb(Whh, lane, HIDDEN, 2 * HIDDEN + jbase, kb);
                v16h b3 = load_b_glb(Whh, lane, HIDDEN, 3 * HIDDEN + jbase, kb);
                v16h a0 = load_a_lds(hcur, lane, HSTR, kb);
                v16h a1 = load_a_lds(hcur + 16 * HSTR, lane, HSTR, kb);
                acc[0][0] = wmma_f16(a0, b0, acc[0][0]);
                acc[0][1] = wmma_f16(a0, b1, acc[0][1]);
                acc[0][2] = wmma_f16(a0, b2, acc[0][2]);
                acc[0][3] = wmma_f16(a0, b3, acc[0][3]);
                acc[1][0] = wmma_f16(a1, b0, acc[1][0]);
                acc[1][1] = wmma_f16(a1, b1, acc[1][1]);
                acc[1][2] = wmma_f16(a1, b2, acc[1][2]);
                acc[1][3] = wmma_f16(a1, b3, acc[1][3]);
            }

            // pointwise cell update; new h goes straight to the other buffer
#pragma unroll
            for (int mt = 0; mt < 2; ++mt)
#pragma unroll
                for (int r = 0; r < 8; ++r) {
                    float iv = sig_fast(acc[mt][0][r] + bia[jj]);
                    float fv = sig_fast(acc[mt][1][r] + bif[jj]);
                    float gv = tanh_fast(acc[mt][2][r] + big[jj]);
                    float ov = sig_fast(acc[mt][3][r] + bio[jj]);
                    float cv = fv * c_reg[jj][mt][r] + iv * gv;
                    c_reg[jj][mt][r] = cv;
                    int mloc = mt * 16 + r + mhalf;
                    hnxt[mloc * HSTR + jbase + ncol] = (_Float16)(ov * tanh_fast(cv));
                }
        }
        __syncthreads();                 // new h complete & visible

        // ---------------- phase 2: out = h @ W_out^T + b_out ----------------
        if (wave < 8) {
            const int mt = wave >> 2;
            const int nt = wave & 3;
            v8f oacc = (v8f){};
            const _Float16* hrow = hnxt + mt * 16 * HSTR;
            for (int kt = 0; kt < 16; ++kt) {
                const int kb = kt * 32;
                v16h a = load_a_lds(hrow, lane, HSTR, kb);
                v16h b = load_b_glb(Wout, lane, HIDDEN, nt * 16, kb);
                oacc = wmma_f16(a, b, oacc);
            }
#pragma unroll
            for (int r = 0; r < 8; ++r) {
                float v  = oacc[r] + ob;
                int mloc = mt * 16 + r + mhalf;
                int brow = rowbase + mloc;
                int col  = nt * 16 + ncol;
                out[((size_t)brow * steps + t) * OUTD + col] = v;   // fp32 result
                x_lds[mloc * ISTR + col] = (_Float16)v;             // feed back
            }
        }
        __syncthreads();                 // inp ready for next step
        cur ^= 1;
    }
}

// ---------------------------------------------------------------------------
extern "C" void kernel_launch(void* const* d_in, const int* in_sizes, int n_in,
                              void* d_out, int out_size, void* d_ws, size_t ws_size,
                              hipStream_t stream) {
    (void)in_sizes; (void)n_in; (void)out_size; (void)ws_size;
    // inputs: x, W_ih, W_hh, b_lstm, h0, c0, W_out, b_out, steps
    const float* W_ih   = (const float*)d_in[1];
    const float* W_hh   = (const float*)d_in[2];
    const float* b_lstm = (const float*)d_in[3];
    const float* h0     = (const float*)d_in[4];
    const float* c0     = (const float*)d_in[5];
    const float* W_out  = (const float*)d_in[6];
    const float* b_out  = (const float*)d_in[7];
    const int*   steps  = (const int*)d_in[8];
    _Float16* wbuf = (_Float16*)d_ws;
    float*    out  = (float*)d_out;

    lstm_convert_weights<<<512, 256, 0, stream>>>(W_ih, W_hh, W_out, wbuf);
    lstm_gen_kernel<<<NBLOCKS, THREADS, 0, stream>>>(b_lstm, h0, c0, b_out, wbuf, steps, out);
}
